// gat_45423574123072
// MI455X (gfx1250) — matrix operand order repaired and checked by reading, not compiled, not run on hardware
//
#include <hip/hip_runtime.h>
#include <hip/hip_bf16.h>
#include <math.h>

typedef __attribute__((ext_vector_type(16))) _Float16 v16h;
typedef __attribute__((ext_vector_type(8)))  _Float16 v8h;
typedef __attribute__((ext_vector_type(8)))  float    v8f;
typedef __attribute__((ext_vector_type(4)))  float    f4v;

#define DIM 512
#define C1  32      // H1*D1
#define H1  4
#define D1  8
#define H2  1
#define D2  2
#define NEG_SLOPE 0.2f
#define LDSP 520    // padded halves per column row: 1040B stride -> bank-conflict free

// Monotone float -> int encoding so signed atomicMax orders like float compare.
__device__ __forceinline__ int enc_f32(float f) {
    int i = __float_as_int(f);
    return i ^ ((i >> 31) & 0x7fffffff);
}
__device__ __forceinline__ float dec_f32(int y) {
    return __int_as_float(y ^ ((y >> 31) & 0x7fffffff));
}

// ---------------------------------------------------------------------------
// Kernel 1: h1 = x @ W1   (WMMA f32_16x16x32_f16, one 16x16 tile per wave)
// W1 staged once per block into LDS as f16, transposed+padded, so each
// K-step's B fragment is two conflict-free ds_load_b128s.
// ---------------------------------------------------------------------------
__global__ void __launch_bounds__(256)
gemm1_wmma(const float* __restrict__ x,
           const float* __restrict__ W1,
           float* __restrict__ h1,
           int n, int n_tiles_total) {
    __shared__ _Float16 sW[C1 * LDSP];   // 33.3 KB

    // ---- cooperative stage: W1[512][32] f32 -> sW[col][k] f16 (transposed) ----
    {
        const f4v* __restrict__ w4 = (const f4v*)W1;
        for (int idx = threadIdx.x; idx < (DIM * C1) / 4; idx += 256) {
            f4v v = w4[idx];
            const int base = idx * 4;
            const int k = base >> 5;        // /32
            const int c = base & 31;        // %32
            sW[(c + 0) * LDSP + k] = (_Float16)v.x;
            sW[(c + 1) * LDSP + k] = (_Float16)v.y;
            sW[(c + 2) * LDSP + k] = (_Float16)v.z;
            sW[(c + 3) * LDSP + k] = (_Float16)v.w;
        }
    }
    __syncthreads();

    const int wave = threadIdx.x >> 5;
    const int lane = threadIdx.x & 31;
    const int t = blockIdx.x * 8 + wave;
    if (t >= n_tiles_total) return;
    const int mt = t >> 1;       // M tile (16 rows)
    const int nt = t & 1;        // N tile (16 of 32 cols)

    const int lh  = lane >> 4;   // lane-half select
    const int l15 = lane & 15;

    int arow = mt * 16 + l15;
    if (arow >= n) arow = n - 1;               // clamp (stores guarded below)
    const float* __restrict__ xrow = x + (size_t)arow * DIM;
    const int bcol = nt * 16 + l15;
    const _Float16* __restrict__ bbase = sW + (size_t)bcol * LDSP;

    v8f acc = {};
    #pragma unroll 4
    for (int k = 0; k < DIM; k += 32) {
        // A tile 16x32 f16 (ISA layout): lane-half 0 -> K {0..7,16..23},
        // lane-half 1 -> K {8..15,24..31}, row = lane&15.
        const int ka = k + lh * 8;
        f4v a0 = *(const f4v*)(xrow + ka + 0);
        f4v a1 = *(const f4v*)(xrow + ka + 4);
        f4v a2 = *(const f4v*)(xrow + ka + 16);
        f4v a3 = *(const f4v*)(xrow + ka + 20);
        v16h A;
        A[0]=(_Float16)a0.x;  A[1]=(_Float16)a0.y;  A[2]=(_Float16)a0.z;  A[3]=(_Float16)a0.w;
        A[4]=(_Float16)a1.x;  A[5]=(_Float16)a1.y;  A[6]=(_Float16)a1.z;  A[7]=(_Float16)a1.w;
        A[8]=(_Float16)a2.x;  A[9]=(_Float16)a2.y;  A[10]=(_Float16)a2.z; A[11]=(_Float16)a2.w;
        A[12]=(_Float16)a3.x; A[13]=(_Float16)a3.y; A[14]=(_Float16)a3.z; A[15]=(_Float16)a3.w;

        // B tile 32x16 f16: lane-half 0 -> K {0..7,16..23}, half 1 -> K {8..15,24..31},
        // col = lane&15. Transposed LDS makes this 2x ds_load_b128.
        const int kb = k + lh * 8;
        const v8h* __restrict__ bp = (const v8h*)(bbase + kb);
        v8h b0 = bp[0];
        v8h b1 = bp[2];
        v16h B;
        #pragma unroll
        for (int j = 0; j < 8; ++j) { B[j] = b0[j]; B[8 + j] = b1[j]; }

        acc = __builtin_amdgcn_wmma_f32_16x16x32_f16(
                  /*neg_a=*/false, A, /*neg_b=*/false, B,
                  /*c_mod=*/(short)0, acc, /*reuse_a=*/false, /*reuse_b=*/false);
    }

    // C layout: VGPR r holds row r + 8*lane_half, col = lane&15.
    float* __restrict__ obase = h1 + ((size_t)(mt * 16 + lh * 8)) * C1 + bcol;
    if (mt * 16 + 16 <= n) {
        #pragma unroll
        for (int r = 0; r < 8; ++r) obase[(size_t)r * C1] = acc[r];
    } else {
        #pragma unroll
        for (int r = 0; r < 8; ++r) {
            const int row = mt * 16 + lh * 8 + r;
            if (row < n) h1[(size_t)row * C1 + bcol] = acc[r];
        }
    }
}

// ---------------------------------------------------------------------------
// Kernel 2: per-node attention logits for conv1
// ---------------------------------------------------------------------------
__global__ void node_attn1(const float* __restrict__ h1,
                           const float* __restrict__ att_s,
                           const float* __restrict__ att_d,
                           float* __restrict__ as1, float* __restrict__ ad1, int n) {
    const int i = blockIdx.x * blockDim.x + threadIdx.x;
    if (i >= n) return;
    const f4v* hr = (const f4v*)(h1 + (size_t)i * C1);
    float s[H1], d[H1];
    #pragma unroll
    for (int h = 0; h < H1; ++h) { s[h] = 0.f; d[h] = 0.f; }
    #pragma unroll
    for (int q = 0; q < C1 / 4; ++q) {
        const f4v v = hr[q];
        const int h = q >> 1;              // D1==8 -> 2 quads per head
        const f4v ws = *(const f4v*)(att_s + q * 4);
        const f4v wd = *(const f4v*)(att_d + q * 4);
        s[h] += v.x * ws.x + v.y * ws.y + v.z * ws.z + v.w * ws.w;
        d[h] += v.x * wd.x + v.y * wd.y + v.z * wd.z + v.w * wd.w;
    }
    #pragma unroll
    for (int h = 0; h < H1; ++h) {
        as1[(size_t)i * H1 + h] = s[h];
        ad1[(size_t)i * H1 + h] = d[h];
    }
}

// ---------------------------------------------------------------------------
// Fill helpers
// ---------------------------------------------------------------------------
__global__ void fill_f32(float* p, float v, long long n) {
    long long i = (long long)blockIdx.x * blockDim.x + threadIdx.x;
    if (i < n) p[i] = v;
}
__global__ void fill_i32(int* p, int v, long long n) {
    long long i = (long long)blockIdx.x * blockDim.x + threadIdx.x;
    if (i < n) p[i] = v;
}

// ---------------------------------------------------------------------------
// Edge passes (segment softmax + aggregate). Self-loops handled implicitly.
// ---------------------------------------------------------------------------
template<int H>
__global__ void edge_max(const int* __restrict__ src, const int* __restrict__ dst,
                         long long E, int n,
                         const float* __restrict__ as, const float* __restrict__ ad,
                         int* __restrict__ menc) {
    long long e = (long long)blockIdx.x * blockDim.x + threadIdx.x;
    if (e >= E + n) return;
    int s, d;
    if (e < E) { s = src[e]; d = dst[e]; } else { s = d = (int)(e - E); }
    #pragma unroll
    for (int h = 0; h < H; ++h) {
        float v = as[(size_t)s * H + h] + ad[(size_t)d * H + h];
        v = v > 0.f ? v : NEG_SLOPE * v;
        atomicMax(&menc[(size_t)d * H + h], enc_f32(v));
    }
}

template<int H>
__global__ void edge_sum(const int* __restrict__ src, const int* __restrict__ dst,
                         long long E, int n,
                         const float* __restrict__ as, const float* __restrict__ ad,
                         const int* __restrict__ menc, float* __restrict__ den) {
    long long e = (long long)blockIdx.x * blockDim.x + threadIdx.x;
    if (e >= E + n) return;
    int s, d;
    if (e < E) { s = src[e]; d = dst[e]; } else { s = d = (int)(e - E); }
    #pragma unroll
    for (int h = 0; h < H; ++h) {
        float v = as[(size_t)s * H + h] + ad[(size_t)d * H + h];
        v = v > 0.f ? v : NEG_SLOPE * v;
        const float m = dec_f32(menc[(size_t)d * H + h]);
        atomicAdd(&den[(size_t)d * H + h], __expf(v - m));
    }
}

template<int H, int D>
__global__ void edge_agg(const int* __restrict__ src, const int* __restrict__ dst,
                         long long E, int n,
                         const float* __restrict__ as, const float* __restrict__ ad,
                         const int* __restrict__ menc, const float* __restrict__ den,
                         const float* __restrict__ hin, float* __restrict__ out) {
    long long e = (long long)blockIdx.x * blockDim.x + threadIdx.x;
    if (e >= E + n) return;
    int s, d;
    if (e < E) { s = src[e]; d = dst[e]; } else { s = d = (int)(e - E); }

    const float* hs = hin + (size_t)s * (H * D);
    __builtin_prefetch(hs, 0, 1);   // global_prefetch_b8 of the message row

    float alpha[H];
    #pragma unroll
    for (int h = 0; h < H; ++h) {
        float v = as[(size_t)s * H + h] + ad[(size_t)d * H + h];
        v = v > 0.f ? v : NEG_SLOPE * v;
        const float m  = dec_f32(menc[(size_t)d * H + h]);
        const float ex = __expf(v - m);
        alpha[h] = ex / (den[(size_t)d * H + h] + 1e-16f);
    }

    // gather message row (vectorized to global_load_b128 when possible)
    float hv[H * D];
    if ((H * D) % 4 == 0) {
        const f4v* hp = (const f4v*)hs;
        #pragma unroll
        for (int q = 0; q < (H * D) / 4; ++q) {
            const f4v v = hp[q];
            hv[q * 4 + 0] = v.x; hv[q * 4 + 1] = v.y;
            hv[q * 4 + 2] = v.z; hv[q * 4 + 3] = v.w;
        }
    } else {
        #pragma unroll
        for (int c = 0; c < H * D; ++c) hv[c] = hs[c];
    }

    float* od = out + (size_t)d * (H * D);
    #pragma unroll
    for (int h = 0; h < H; ++h)
        #pragma unroll
        for (int j = 0; j < D; ++j)
            atomicAdd(&od[h * D + j], alpha[h] * hv[h * D + j]);
}

// ---------------------------------------------------------------------------
// Kernel: o1 = relu(agg1 + b1); h2 = o1 @ W2; attention logits for conv2
// ---------------------------------------------------------------------------
__global__ void node_fuse2(const float* __restrict__ agg1, const float* __restrict__ b1,
                           const float* __restrict__ W2,
                           const float* __restrict__ att_s2, const float* __restrict__ att_d2,
                           float* __restrict__ h2, float* __restrict__ as2,
                           float* __restrict__ ad2, int n) {
    const int i = blockIdx.x * blockDim.x + threadIdx.x;
    if (i >= n) return;
    const f4v* ar = (const f4v*)(agg1 + (size_t)i * C1);
    float hv0 = 0.f, hv1 = 0.f;
    #pragma unroll
    for (int q = 0; q < C1 / 4; ++q) {
        const f4v a = ar[q];
        const f4v bb = *(const f4v*)(b1 + q * 4);
        float o0 = a.x + bb.x; o0 = o0 > 0.f ? o0 : 0.f;
        float o1 = a.y + bb.y; o1 = o1 > 0.f ? o1 : 0.f;
        float o2 = a.z + bb.z; o2 = o2 > 0.f ? o2 : 0.f;
        float o3 = a.w + bb.w; o3 = o3 > 0.f ? o3 : 0.f;
        const int c = q * 4;
        hv0 += o0 * W2[(c+0)*D2] + o1 * W2[(c+1)*D2] + o2 * W2[(c+2)*D2] + o3 * W2[(c+3)*D2];
        hv1 += o0 * W2[(c+0)*D2+1] + o1 * W2[(c+1)*D2+1] + o2 * W2[(c+2)*D2+1] + o3 * W2[(c+3)*D2+1];
    }
    h2[(size_t)i * D2 + 0] = hv0;
    h2[(size_t)i * D2 + 1] = hv1;
    as2[i] = hv0 * att_s2[0] + hv1 * att_s2[1];
    ad2[i] = hv0 * att_d2[0] + hv1 * att_d2[1];
}

__global__ void add_bias2(float* __restrict__ out, const float* __restrict__ b2, int n) {
    const int i = blockIdx.x * blockDim.x + threadIdx.x;
    if (i >= n * D2) return;
    out[i] += b2[i % D2];
}

// ---------------------------------------------------------------------------
// Host-side orchestration
// ---------------------------------------------------------------------------
extern "C" void kernel_launch(void* const* d_in, const int* in_sizes, int n_in,
                              void* d_out, int out_size, void* d_ws, size_t ws_size,
                              hipStream_t stream) {
    const float* x        = (const float*)d_in[0];
    const int*   ei       = (const int*)d_in[1];
    const float* W1       = (const float*)d_in[2];
    const float* att_src1 = (const float*)d_in[3];
    const float* att_dst1 = (const float*)d_in[4];
    const float* b1       = (const float*)d_in[5];
    const float* W2       = (const float*)d_in[6];
    const float* att_src2 = (const float*)d_in[7];
    const float* att_dst2 = (const float*)d_in[8];
    const float* b2       = (const float*)d_in[9];
    float* out = (float*)d_out;

    const int       N  = in_sizes[0] / DIM;
    const long long E  = (long long)in_sizes[1] / 2;
    const long long Et = E + N;
    const int* src = ei;
    const int* dst = ei + E;

    // Workspace carve-out (floats unless noted)
    float* h1   = (float*)d_ws;             // N*32
    float* as1  = h1   + (size_t)N * C1;    // N*4
    float* ad1  = as1  + (size_t)N * H1;    // N*4
    int*   m1   = (int*)(ad1 + (size_t)N * H1);       // N*4 (encoded)
    float* den1 = (float*)(m1 + (size_t)N * H1);      // N*4
    float* agg1 = den1 + (size_t)N * H1;    // N*32
    float* h2   = agg1 + (size_t)N * C1;    // N*2
    float* as2  = h2   + (size_t)N * D2;    // N
    float* ad2  = as2  + (size_t)N;         // N
    int*   m2   = (int*)(ad2 + (size_t)N);  // N (encoded)
    float* den2 = (float*)(m2 + (size_t)N); // N

    const int ENC_NEG_INF = (int)0x807FFFFF;  // enc_f32(-inf)

    const int TB = 256;
    const int nodeBlocks = (N + TB - 1) / TB;
    const int edgeBlocks = (int)((Et + TB - 1) / TB);
    auto fillBlocks = [&](long long cnt) { return (int)((cnt + TB - 1) / TB); };

    // ---- conv1 ----
    {
        const int tiles = ((N + 15) / 16) * 2;
        const int blocks = (tiles + 7) / 8;    // 8 waves per 256-thread block
        gemm1_wmma<<<blocks, TB, 0, stream>>>(x, W1, h1, N, tiles);
    }
    node_attn1<<<nodeBlocks, TB, 0, stream>>>(h1, att_src1, att_dst1, as1, ad1, N);

    fill_i32<<<fillBlocks((long long)N * H1), TB, 0, stream>>>(m1,   ENC_NEG_INF, (long long)N * H1);
    fill_f32<<<fillBlocks((long long)N * H1), TB, 0, stream>>>(den1, 0.f,         (long long)N * H1);
    fill_f32<<<fillBlocks((long long)N * C1), TB, 0, stream>>>(agg1, 0.f,         (long long)N * C1);

    edge_max<H1><<<edgeBlocks, TB, 0, stream>>>(src, dst, E, N, as1, ad1, m1);
    edge_sum<H1><<<edgeBlocks, TB, 0, stream>>>(src, dst, E, N, as1, ad1, m1, den1);
    edge_agg<H1, D1><<<edgeBlocks, TB, 0, stream>>>(src, dst, E, N, as1, ad1, m1, den1, h1, agg1);

    // ---- conv2 ----
    node_fuse2<<<nodeBlocks, TB, 0, stream>>>(agg1, b1, W2, att_src2, att_dst2,
                                              h2, as2, ad2, N);

    fill_i32<<<fillBlocks(N), TB, 0, stream>>>(m2,   ENC_NEG_INF, (long long)N);
    fill_f32<<<fillBlocks(N), TB, 0, stream>>>(den2, 0.f,         (long long)N);
    fill_f32<<<fillBlocks((long long)N * D2), TB, 0, stream>>>(out, 0.f, (long long)N * D2);

    edge_max<H2><<<edgeBlocks, TB, 0, stream>>>(src, dst, E, N, as2, ad2, m2);
    edge_sum<H2><<<edgeBlocks, TB, 0, stream>>>(src, dst, E, N, as2, ad2, m2, den2);
    edge_agg<H2, D2><<<edgeBlocks, TB, 0, stream>>>(src, dst, E, N, as2, ad2, m2, den2, h2, out);

    add_bias2<<<(N * D2 + TB - 1) / TB, TB, 0, stream>>>(out, b2, N);
}
